// dgl_MPNNGNN_39625368273427
// MI455X (gfx1250) — matrix operand order repaired and checked by reading, not compiled, not run on hardware
//
#include <hip/hip_runtime.h>

// ---------------------------------------------------------------------------
// MPNN-GNN forward for MI455X (gfx1250, wave32, WMMA + TDM).
//
// Strategy: recompute edge weight matrices per step inside the message kernel
// with bf16 WMMA (16x16x32, f32 accum) instead of materializing the ~1 GB
// ew tensor; everything hot stays L2-resident. The per-block A-panel is
// staged into LDS by the Tensor Data Mover, using TDM pad-insertion to
// produce the 272B padded row pitch directly.
// ---------------------------------------------------------------------------

typedef __bf16 bf16;
typedef __attribute__((ext_vector_type(16))) __bf16 v16bf;
typedef __attribute__((ext_vector_type(8)))  float  v8f;
typedef __attribute__((ext_vector_type(4))) unsigned int u32x4;
typedef __attribute__((ext_vector_type(8))) int i32x8;
typedef __attribute__((ext_vector_type(4))) int i32x4;

#define N_NODES 15000
#define NPAD    15008          // 938 * 16
#define N_EDGES 60000
#define D       64
#define DEH     128
#define DNODE   74
#define DEDGE   12
#define G3D     192            // 3*D (GRU gates)

#ifdef __has_builtin
#  if __has_builtin(__builtin_amdgcn_tensor_load_to_lds)
#    define USE_TDM 1
#  endif
#endif
#ifndef USE_TDM
#  define USE_TDM 0
#endif

__device__ static inline v8f zero8() {
  v8f z;
#pragma unroll
  for (int i = 0; i < 8; ++i) z[i] = 0.0f;
  return z;
}

// A-matrix fragment, 16x32 bf16, row-major source with leading dim `lda`
// (elements). ISA layout: lanes 0-15 -> M=lane, Kbase=0; lanes 16-31 ->
// M=lane-16, Kbase=8. frag elems 0..7 = K[Kbase..Kbase+7], elems 8..15 =
// K[16+Kbase..16+Kbase+7]. Two aligned 16B loads per lane.
__device__ static inline v16bf load_a_frag(const bf16* base, int lda, int lane) {
  int m  = lane & 15;
  int kb = (lane & 16) ? 8 : 0;
  const bf16* p = base + (size_t)m * lda + kb;
  union { v16bf v; uint4 q[2]; } u;
  u.q[0] = *(const uint4*)(p);
  u.q[1] = *(const uint4*)(p + 16);
  return u.v;
}

// B-matrix fragment, 32x16 bf16, from TRANSPOSED storage bt[n][k] with leading
// dim `ldb`. ISA layout: lane -> N=lane&15; Kbase=0 (lanes 0-15) or 8 (16-31);
// elems 0..7 = K[Kbase..Kbase+7], elems 8..15 = K[16+Kbase..]. Two uint4.
__device__ static inline v16bf load_b_frag_t(const bf16* baseT, int ldb, int lane) {
  int n  = lane & 15;
  int kb = (lane & 16) ? 8 : 0;
  const bf16* p = baseT + (size_t)n * ldb + kb;
  union { v16bf v; uint4 q[2]; } u;
  u.q[0] = *(const uint4*)(p);
  u.q[1] = *(const uint4*)(p + 16);
  return u.v;
}

__device__ static inline v8f wmma_bf16(v16bf a, v16bf b, v8f c) {
  return __builtin_amdgcn_wmma_f32_16x16x32_bf16(false, a, false, b,
                                                 (short)0, c, false, false);
}

__device__ static inline float sigmoidf(float x) { return 1.0f / (1.0f + expf(-x)); }

// ---------------------------------------------------------------------------
// One-time prep: W_e2 [128,4096] -> bf16 transposed W2t[4096][128];
// W_ih/W_hh [192,64] -> bf16 (row-major already equals transposed-B layout).
// ---------------------------------------------------------------------------
__global__ void prep_kernel(const float* __restrict__ We2,
                            const float* __restrict__ Wih,
                            const float* __restrict__ Whh,
                            bf16* __restrict__ W2t,
                            bf16* __restrict__ WihB,
                            bf16* __restrict__ WhhB) {
  int tid  = blockIdx.x * blockDim.x + threadIdx.x;
  int nthr = gridDim.x * blockDim.x;
  for (int idx = tid; idx < 4096 * DEH; idx += nthr) {
    int n = idx / DEH, k = idx % DEH;
    W2t[idx] = (bf16)We2[(size_t)k * 4096 + n];
  }
  for (int idx = tid; idx < G3D * D; idx += nthr) WihB[idx] = (bf16)Wih[idx];
  for (int idx = tid; idx < G3D * D; idx += nthr) WhhB[idx] = (bf16)Whh[idx];
}

__global__ void zero_f32_kernel(float* __restrict__ p, int n) {
  int i = blockIdx.x * blockDim.x + threadIdx.x;
  if (i < n) p[i] = 0.0f;
}
__global__ void zero_i32_kernel(int* __restrict__ p, int n) {
  int i = blockIdx.x * blockDim.x + threadIdx.x;
  if (i < n) p[i] = 0;
}
__global__ void deg_kernel(const int* __restrict__ dst, int* __restrict__ deg) {
  int e = blockIdx.x * blockDim.x + threadIdx.x;
  if (e < N_EDGES) atomicAdd(&deg[dst[e]], 1);
}
__global__ void invdeg_kernel(const int* __restrict__ deg, float* __restrict__ inv) {
  int n = blockIdx.x * blockDim.x + threadIdx.x;
  if (n < NPAD) {
    int dg = (n < N_NODES) ? deg[n] : 0;
    inv[n] = (dg > 0) ? (1.0f / (float)dg) : 0.0f;
  }
}

// h = relu(node_feats @ W_proj + b_proj); pad rows zeroed.
__global__ __launch_bounds__(64)
void node_proj_kernel(const float* __restrict__ nf, const float* __restrict__ Wp,
                      const float* __restrict__ bp, float* __restrict__ hidden) {
  __shared__ float f[DNODE];
  int n = blockIdx.x, d = threadIdx.x;
  if (n >= N_NODES) { hidden[(size_t)n * D + d] = 0.0f; return; }
  for (int i = d; i < DNODE; i += 64) f[i] = nf[(size_t)n * DNODE + i];
  __syncthreads();
  float acc = bp[d];
#pragma unroll 2
  for (int i = 0; i < DNODE; ++i) acc += f[i] * Wp[i * D + d];
  hidden[(size_t)n * D + d] = acc > 0.0f ? acc : 0.0f;
}

// a = bf16(relu(edge_feats @ W_e1 + b_e1))
__global__ __launch_bounds__(128)
void edge_hidden_kernel(const float* __restrict__ ef, const float* __restrict__ W1,
                        const float* __restrict__ b1, bf16* __restrict__ aW) {
  __shared__ float f[DEDGE];
  int e = blockIdx.x, k = threadIdx.x;
  if (k < DEDGE) f[k] = ef[(size_t)e * DEDGE + k];
  __syncthreads();
  float acc = b1[k];
#pragma unroll
  for (int i = 0; i < DEDGE; ++i) acc += f[i] * W1[i * DEH + k];
  acc = acc > 0.0f ? acc : 0.0f;
  aW[(size_t)e * DEH + k] = (bf16)acc;
}

// ---------------------------------------------------------------------------
// Message kernel (the hot loop): per block 32 edges; 8 waves = 2 edge-groups
// x 4 output-column tiles. For each of the 64 input channels i, each wave
// recomputes a 16x16 tile of ew = a @ W_e2 with four K=32 bf16 WMMAs,
// multiplies by gathered h[src][i], and accumulates the message; finally
// scatters with global f32 atomics into rst[dst].
//
// The 32x128 bf16 A-panel is staged into LDS by the Tensor Data Mover: a 2-D
// D# (tile 128x32, data_size=2B) with pad-insertion (4 DWORDs after every
// 64 DWORDs) produces the 272-byte padded LDS row pitch in hardware.
// ---------------------------------------------------------------------------
__global__ __launch_bounds__(256)
void msg_kernel(const bf16* __restrict__ aW,     // [E][128]
                const bf16* __restrict__ W2t,    // [4096][128] (transposed)
                const float* __restrict__ b2,    // [4096]
                const float* __restrict__ hidden,// [NPAD][64]
                const int* __restrict__ src, const int* __restrict__ dst,
                float* __restrict__ rst) {
  __shared__ __align__(16) bf16 aS[32][136];     // 128 + 8 pad -> 272B rows
  __shared__ float hS[32][D];
  __shared__ int srcS[32], dstS[32];

  int t  = threadIdx.x;
  int e0 = blockIdx.x * 32;

  if (t < 32) { srcS[t] = src[e0 + t]; dstS[t] = dst[e0 + t]; }

#if USE_TDM
  if (t < 32) {   // wave 0 issues the TDM copy (EXEC-independent, wave-scalar)
    unsigned int lds_addr = (unsigned int)(size_t)&aS[0][0];
    unsigned long long gaddr = (unsigned long long)(size_t)(aW + (size_t)e0 * DEH);
    u32x4 g0;
    g0[0] = 1u;                               // count=1 (valid descriptor)
    g0[1] = lds_addr;                         // LDS byte address
    g0[2] = (unsigned int)gaddr;              // global addr [31:0]
    g0[3] = (unsigned int)((gaddr >> 32) & 0x01FFFFFFu) | 0x80000000u; // type=2
    i32x8 g1;
    g1[0] = (1 << 16)                         // data_size = 2 bytes
          | (1 << 20)                         // pad_enable
          | (5 << 22)                         // pad_interval: 64 DWORDs (1 row)
          | (3 << 25);                        // pad_amount: 4 DWORDs (16B)
    g1[1] = DEH << 16;                        // tensor_dim0 = 128  [63:48]
    g1[2] = 32 << 16;                         // tensor_dim1 = 32   [95:80]
    g1[3] = DEH << 16;                        // tile_dim0 = 128    [127:112]
    g1[4] = 32;                               // tile_dim1 = 32     [143:128]
    g1[5] = DEH;                              // tensor_dim0_stride [191:160]
    g1[6] = 0;
    g1[7] = 0;
    i32x4 gz;
    gz[0] = 0; gz[1] = 0; gz[2] = 0; gz[3] = 0;
#if __clang_major__ >= 23
    i32x8 gz8;
#pragma unroll
    for (int i = 0; i < 8; ++i) gz8[i] = 0;
    __builtin_amdgcn_tensor_load_to_lds(g0, g1, gz, gz, gz8, 0);
#else
    __builtin_amdgcn_tensor_load_to_lds(g0, g1, gz, gz, 0);
#endif
    __builtin_amdgcn_s_wait_tensorcnt(0);
  }
#else
  {   // fallback: manual staging, 16 bf16 per thread
    int el = t >> 3;
    int k0 = (t & 7) * 16;
    const uint4* g = (const uint4*)(aW + (size_t)(e0 + el) * DEH + k0);
    uint4 q0 = g[0], q1 = g[1];
    *(uint4*)&aS[el][k0]     = q0;
    *(uint4*)&aS[el][k0 + 8] = q1;
  }
#endif
  __syncthreads();
  for (int idx = t; idx < 32 * D; idx += 256) {   // gather h[src] rows
    int el = idx >> 6, i = idx & 63;
    hS[el][i] = hidden[(size_t)srcS[el] * D + i];
  }
  __syncthreads();

  int w = t >> 5, lane = t & 31;
  int g  = w >> 2;           // which 16-edge group
  int o0 = (w & 3) * 16;     // output column tile

  v16bf af[4];
#pragma unroll
  for (int kc = 0; kc < 4; ++kc)
    af[kc] = load_a_frag(&aS[g * 16][0] + kc * 32, 136, lane);

  const int rowoff = (lane & 16) ? 8 : 0;
  const int col    = lane & 15;
  v8f msgAcc = zero8();

  for (int i = 0; i < D; ++i) {
    v8f ew = zero8();
    const bf16* bbase = W2t + (size_t)(i * D + o0) * DEH;
#pragma unroll
    for (int kc = 0; kc < 4; ++kc)
      ew = wmma_bf16(af[kc], load_b_frag_t(bbase + kc * 32, DEH, lane), ew);
    float b2v = b2[i * D + o0 + col];
#pragma unroll
    for (int r = 0; r < 8; ++r) {
      float hv = hS[g * 16 + r + rowoff][i];   // LDS broadcast per half-wave
      msgAcc[r] += hv * (ew[r] + b2v);
    }
  }
#pragma unroll
  for (int r = 0; r < 8; ++r) {
    int dn = dstS[g * 16 + r + rowoff];
    atomicAdd(&rst[(size_t)dn * D + o0 + col], msgAcc[r]);
  }
}

// rst = rst*inv_deg + conv_bias, plus BN sum / sum-of-squares partials.
__global__ __launch_bounds__(256)
void stats_kernel(float* __restrict__ rst, const float* __restrict__ inv_deg,
                  const float* __restrict__ conv_bias, float* __restrict__ bnsum) {
  int tid    = blockIdx.x * blockDim.x + threadIdx.x;
  int nthr   = gridDim.x * blockDim.x;
  int d      = tid & 63;
  int stride = nthr >> 6;
  float cb = conv_bias[d];
  float s = 0.0f, s2 = 0.0f;
  for (int n = tid >> 6; n < N_NODES; n += stride) {
    size_t idx = (size_t)n * D + d;
    float v = rst[idx] * inv_deg[n] + cb;
    rst[idx] = v;
    s += v; s2 += v * v;
  }
  atomicAdd(&bnsum[d], s);
  atomicAdd(&bnsum[64 + d], s2);
}

// Fold BN into per-channel scale/shift: m = relu(rst*scl + sft)
__global__ void bn_finalize_kernel(const float* __restrict__ bnsum,
                                   const float* __restrict__ gamma,
                                   const float* __restrict__ beta,
                                   float* __restrict__ bnpar) {
  int d = threadIdx.x;  // 64 threads
  const float invN = 1.0f / (float)N_NODES;
  float mu   = bnsum[d] * invN;
  float var  = bnsum[64 + d] * invN - mu * mu;
  float rsig = rsqrtf(var + 1e-5f);
  float scl  = rsig * gamma[d];
  bnpar[d]      = scl;
  bnpar[64 + d] = beta[d] - mu * scl;
}

// ---------------------------------------------------------------------------
// Fused BN + ReLU + GRU cell. 16 nodes per block, 4 waves. Two [16,64]@[64,192]
// bf16 WMMA GEMMs (gi = m@Wih^T, gh = hidden@Whh^T), fp32 gates in LDS.
// ---------------------------------------------------------------------------
__global__ __launch_bounds__(128)
void bn_gru_kernel(const float* __restrict__ rst,
                   const float* __restrict__ bnpar,
                   float* __restrict__ hidden,       // in/out
                   const bf16* __restrict__ WihB,    // [192][64] bf16
                   const bf16* __restrict__ WhhB,
                   const float* __restrict__ b_ih, const float* __restrict__ b_hh,
                   float* __restrict__ out, int lastFlag) {
  __shared__ __align__(16) bf16 mS[16][72];   // 144B rows (16B multiple)
  __shared__ __align__(16) bf16 hSb[16][72];
  __shared__ float hF[16][D];
  __shared__ float GI[16][G3D];
  __shared__ float GH[16][G3D];

  int t  = threadIdx.x;
  int nb = blockIdx.x * 16;

  for (int idx = t; idx < 16 * D; idx += 128) {
    int nl = idx >> 6, dd = idx & 63;
    float v = rst[(size_t)(nb + nl) * D + dd];
    float m = v * bnpar[dd] + bnpar[64 + dd];
    m = m > 0.0f ? m : 0.0f;
    mS[nl][dd] = (bf16)m;
    float hv = hidden[(size_t)(nb + nl) * D + dd];
    hF[nl][dd]  = hv;
    hSb[nl][dd] = (bf16)hv;
  }
  __syncthreads();

  int w = t >> 5, lane = t & 31;
  v16bf mf0 = load_a_frag(&mS[0][0],       72, lane);
  v16bf mf1 = load_a_frag(&mS[0][0] + 32,  72, lane);
  v16bf hf0 = load_a_frag(&hSb[0][0],      72, lane);
  v16bf hf1 = load_a_frag(&hSb[0][0] + 32, 72, lane);
  int col = lane & 15, rowoff = (lane & 16) ? 8 : 0;

#pragma unroll
  for (int tt = 0; tt < 3; ++tt) {
    int n0 = w * 48 + tt * 16;
    v8f gi = zero8();
    gi = wmma_bf16(mf0, load_b_frag_t(WihB + (size_t)n0 * D,      D, lane), gi);
    gi = wmma_bf16(mf1, load_b_frag_t(WihB + (size_t)n0 * D + 32, D, lane), gi);
    v8f gh = zero8();
    gh = wmma_bf16(hf0, load_b_frag_t(WhhB + (size_t)n0 * D,      D, lane), gh);
    gh = wmma_bf16(hf1, load_b_frag_t(WhhB + (size_t)n0 * D + 32, D, lane), gh);
    float bi = b_ih[n0 + col];
    float bh = b_hh[n0 + col];
#pragma unroll
    for (int r = 0; r < 8; ++r) {
      GI[r + rowoff][n0 + col] = gi[r] + bi;
      GH[r + rowoff][n0 + col] = gh[r] + bh;
    }
  }
  __syncthreads();

  for (int idx = t; idx < 16 * D; idx += 128) {
    int nl = idx >> 6, dd = idx & 63;
    float rr = sigmoidf(GI[nl][dd]       + GH[nl][dd]);
    float zz = sigmoidf(GI[nl][64 + dd]  + GH[nl][64 + dd]);
    float nn = tanhf  (GI[nl][128 + dd] + rr * GH[nl][128 + dd]);
    float hnew = (1.0f - zz) * nn + zz * hF[nl][dd];
    int n = nb + nl;
    if (n < N_NODES) {
      hidden[(size_t)n * D + dd] = hnew;
      if (lastFlag) out[(size_t)n * D + dd] = hnew;
    }
  }
}

// ---------------------------------------------------------------------------
extern "C" void kernel_launch(void* const* d_in, const int* in_sizes, int n_in,
                              void* d_out, int out_size, void* d_ws, size_t ws_size,
                              hipStream_t stream) {
  (void)in_sizes; (void)n_in; (void)out_size; (void)ws_size;

  const float* node_feats = (const float*)d_in[0];
  const float* edge_feats = (const float*)d_in[1];
  const int*   src        = (const int*)d_in[2];
  const int*   dst        = (const int*)d_in[3];
  const float* W_proj     = (const float*)d_in[4];
  const float* b_proj     = (const float*)d_in[5];
  const float* W_e1       = (const float*)d_in[6];
  const float* b_e1       = (const float*)d_in[7];
  const float* W_e2       = (const float*)d_in[8];
  const float* b_e2       = (const float*)d_in[9];
  const float* conv_bias  = (const float*)d_in[10];
  const float* bn_gamma   = (const float*)d_in[11];
  const float* bn_beta    = (const float*)d_in[12];
  const float* W_ih       = (const float*)d_in[13];
  const float* W_hh       = (const float*)d_in[14];
  const float* b_ih       = (const float*)d_in[15];
  const float* b_hh       = (const float*)d_in[16];
  float* out = (float*)d_out;

  char* ws = (char*)d_ws;
  auto alloc = [&](size_t bytes) -> char* {
    char* p = ws; ws += (bytes + 255) & ~(size_t)255; return p;
  };
  float* hidden  = (float*)alloc((size_t)NPAD * D * 4);
  float* rst     = (float*)alloc((size_t)NPAD * D * 4);
  bf16*  aW      = (bf16*) alloc((size_t)N_EDGES * DEH * 2);
  bf16*  W2t     = (bf16*) alloc((size_t)4096 * DEH * 2);
  bf16*  WihB    = (bf16*) alloc((size_t)G3D * D * 2);
  bf16*  WhhB    = (bf16*) alloc((size_t)G3D * D * 2);
  float* inv_deg = (float*)alloc((size_t)NPAD * 4);
  int*   deg     = (int*)  alloc((size_t)NPAD * 4);
  float* bnsum   = (float*)alloc(128 * 4);
  float* bnpar   = (float*)alloc(128 * 4);

  prep_kernel<<<512, 256, 0, stream>>>(W_e2, W_ih, W_hh, W2t, WihB, WhhB);
  zero_i32_kernel<<<(NPAD + 255) / 256, 256, 0, stream>>>(deg, NPAD);
  deg_kernel<<<(N_EDGES + 255) / 256, 256, 0, stream>>>(dst, deg);
  invdeg_kernel<<<(NPAD + 255) / 256, 256, 0, stream>>>(deg, inv_deg);
  node_proj_kernel<<<NPAD, 64, 0, stream>>>(node_feats, W_proj, b_proj, hidden);
  edge_hidden_kernel<<<N_EDGES, 128, 0, stream>>>(edge_feats, W_e1, b_e1, aW);

  for (int step = 0; step < 6; ++step) {
    zero_f32_kernel<<<(NPAD * D + 255) / 256, 256, 0, stream>>>(rst, NPAD * D);
    zero_f32_kernel<<<1, 128, 0, stream>>>(bnsum, 128);
    msg_kernel<<<N_EDGES / 32, 256, 0, stream>>>(aW, W2t, b_e2, hidden, src, dst, rst);
    stats_kernel<<<240, 256, 0, stream>>>(rst, inv_deg, conv_bias, bnsum);
    bn_finalize_kernel<<<1, 64, 0, stream>>>(bnsum, bn_gamma, bn_beta, bnpar);
    bn_gru_kernel<<<NPAD / 16, 128, 0, stream>>>(rst, bnpar, hidden, WihB, WhhB,
                                                 b_ih, b_hh, out, step == 5 ? 1 : 0);
  }
}
